// TemporalFeatureExtractor_7464653160721
// MI455X (gfx1250) — compile-verified, matrix operand order checked by reading, not run on hardware
//
#include <hip/hip_runtime.h>
#include <hip/hip_bf16.h>
#include <stdint.h>

// ---------------------------------------------------------------------------
// TemporalFeatureExtractor for MI455X (gfx1250, wave32, WMMA).
//
// Input flattening assumption (setup_inputs() dict order, depth-first):
//  [0] x[32,512,256]
//  layer0: [1]W_i [2]b_i [3]W_f [4]b_f [5]W_o [6]b_o [7]W_g [8]b_g
//          [9]g_c [10]be_c [11]g_h [12]be_h
//  layer1: [13..24] same order (W_* are [512,1024])
//  conv:   [25]w0 [26]b0 [27]w1 [28]b1 [29]w2 [30]b2
//  attn:   [31..34]  (DEAD CODE in reference -> ignored)
//  [35] proj_w[512,2048]  [36] proj_b[512]
//
// d_out = temporal[32,512,512] ++ h_fin[2,32,512] ++ c_fin[2,32,512] (fp32)
// ---------------------------------------------------------------------------

typedef __bf16 bf16_t;
typedef __attribute__((ext_vector_type(16))) __bf16 v16bf;
typedef __attribute__((ext_vector_type(8)))  __bf16 v8bf;
typedef __attribute__((ext_vector_type(8)))  float  v8f;

#define Bk   32
#define Sk   512
#define DINk 256
#define Hk   512
#define SP   520   // S + 8 (4 zero-pad rows each side of time axis)
#define HST  520   // LDS h-state row stride (elems, bank-spread pad)
#define XST  264   // LDS x-stage  row stride (elems)

// Load one 16-element WMMA fragment slice (A or B) from a row-major source.
// Per CDNA5 16-bit operand layout: lanes 0-15 take K = [kb..kb+7, kb+16..kb+23]
// (caller adds koff = (lane>>4)*8 so lanes 16-31 take the other halves).
__device__ __forceinline__ v16bf load_frag16(const bf16_t* rowptr, int kbase) {
  v8bf lo = *(const v8bf*)(rowptr + kbase);
  v8bf hi = *(const v8bf*)(rowptr + kbase + 16);
  v16bf r;
#pragma unroll
  for (int i = 0; i < 8; ++i) { r[i] = lo[i]; r[i + 8] = hi[i]; }
  return r;
}

__device__ __forceinline__ float sigmoidf_(float x) {
  return 1.0f / (1.0f + __expf(-x));
}

// One K-segment of the 4-gate GEMM: A rows from `arow` (segment-local k),
// B rows from Wl at absolute K offset kb0.  No per-iteration branching.
__device__ __forceinline__ void gemm4_seg(const bf16_t* arow, int koff,
                                          const bf16_t* __restrict__ Wl, int K,
                                          int kb0, int nch, int col, v8f* acc) {
  const bf16_t* bp0 = Wl + (size_t)(0 * Hk + col) * K + kb0 + koff;
  const bf16_t* bp1 = Wl + (size_t)(1 * Hk + col) * K + kb0 + koff;
  const bf16_t* bp2 = Wl + (size_t)(2 * Hk + col) * K + kb0 + koff;
  const bf16_t* bp3 = Wl + (size_t)(3 * Hk + col) * K + kb0 + koff;
#pragma unroll 4
  for (int kc = 0; kc < nch; ++kc) {
    const int kb = kc << 5;
    v16bf a = load_frag16(arow, kb + koff);
    __builtin_prefetch(bp0 + kb + 32, 0, 0);
    v16bf b0 = load_frag16(bp0 + kb, 0);
    acc[0] = __builtin_amdgcn_wmma_f32_16x16x32_bf16(false, a, false, b0, (short)0,
                                                     acc[0], false, false);
    __builtin_prefetch(bp1 + kb + 32, 0, 0);
    v16bf b1 = load_frag16(bp1 + kb, 0);
    acc[1] = __builtin_amdgcn_wmma_f32_16x16x32_bf16(false, a, false, b1, (short)0,
                                                     acc[1], false, false);
    __builtin_prefetch(bp2 + kb + 32, 0, 0);
    v16bf b2 = load_frag16(bp2 + kb, 0);
    acc[2] = __builtin_amdgcn_wmma_f32_16x16x32_bf16(false, a, false, b2, (short)0,
                                                     acc[2], false, false);
    __builtin_prefetch(bp3 + kb + 32, 0, 0);
    v16bf b3 = load_frag16(bp3 + kb, 0);
    acc[3] = __builtin_amdgcn_wmma_f32_16x16x32_bf16(false, a, false, b3, (short)0,
                                                     acc[3], false, false);
  }
}

// ---------------------------------------------------------------------------
// Weight prep kernels (fp32 -> bf16, layout changes)
// ---------------------------------------------------------------------------
__global__ void cvt_lstm_w(const float* __restrict__ Wi, const float* __restrict__ Wf,
                           const float* __restrict__ Wo, const float* __restrict__ Wg,
                           const float* __restrict__ bi, const float* __restrict__ bfv,
                           const float* __restrict__ bo, const float* __restrict__ bg,
                           bf16_t* __restrict__ Wcat, float* __restrict__ bcat, int K) {
  size_t total = (size_t)2048 * K;
  for (size_t i = (size_t)blockIdx.x * blockDim.x + threadIdx.x; i < total;
       i += (size_t)gridDim.x * blockDim.x) {
    int n = (int)(i / K);
    int k = (int)(i % K);
    int g = n >> 9, r = n & 511;
    const float* Ws = (g == 0) ? Wi : (g == 1) ? Wf : (g == 2) ? Wo : Wg;
    Wcat[i] = (bf16_t)Ws[(size_t)r * K + k];
    if (k == 0) {
      const float* bs = (g == 0) ? bi : (g == 1) ? bfv : (g == 2) ? bo : bg;
      bcat[n] = bs[r];
    }
  }
}

__global__ void cvt_conv_w(const float* __restrict__ w, bf16_t* __restrict__ wr) {
  // wr[k][o][c] = w[o][c][k];  w is [512][512][3]
  size_t total = (size_t)3 * 512 * 512;
  for (size_t i = (size_t)blockIdx.x * blockDim.x + threadIdx.x; i < total;
       i += (size_t)gridDim.x * blockDim.x) {
    int k = (int)(i / (512 * 512));
    int rem = (int)(i % (512 * 512));
    int o = rem >> 9, c = rem & 511;
    wr[i] = (bf16_t)w[((size_t)o * 512 + c) * 3 + k];
  }
}

__global__ void cvt_f2bf(const float* __restrict__ src, bf16_t* __restrict__ dst, size_t n) {
  for (size_t i = (size_t)blockIdx.x * blockDim.x + threadIdx.x; i < n;
       i += (size_t)gridDim.x * blockDim.x)
    dst[i] = (bf16_t)src[i];
}

__global__ void zero_bf16(bf16_t* __restrict__ p, size_t n) {
  for (size_t i = (size_t)blockIdx.x * blockDim.x + threadIdx.x; i < n;
       i += (size_t)gridDim.x * blockDim.x)
    p[i] = (bf16_t)0.0f;
}

// ---------------------------------------------------------------------------
// LSTM: 2 blocks x 1024 threads; block = 16 batch rows, wave = 16 hidden cols.
// Gates GEMM [16, K] x [K, 2048] via bf16 WMMA; A from LDS (x-stage / h-state),
// B streamed from global bf16 weights. c-state lives in WMMA-D-layout regs.
// K-loop is split per A-source segment so the hot loop has no branches.
// ---------------------------------------------------------------------------
__global__ __launch_bounds__(1024) void lstm_kernel(
    const float* __restrict__ x,
    const bf16_t* __restrict__ Wcat0, const bf16_t* __restrict__ Wcat1,
    const float* __restrict__ bcat,
    const float* __restrict__ gc0, const float* __restrict__ bec0,
    const float* __restrict__ gh0, const float* __restrict__ beh0,
    const float* __restrict__ gc1, const float* __restrict__ bec1,
    const float* __restrict__ gh1, const float* __restrict__ beh1,
    bf16_t* __restrict__ lstm_pad,   // [B][SP][H] bf16
    float* __restrict__ hc_out)      // h_fin[2][B][H] ++ c_fin[2][B][H]
{
  extern __shared__ char smem[];
  bf16_t* hstate = (bf16_t*)smem;                 // [2][16][HST]
  bf16_t* xstage = hstate + 2 * 16 * HST;         // [16][XST]
  float*  red    = (float*)(xstage + 16 * XST);   // [2 layers][4][16]

  const int tid   = threadIdx.x;
  const int lane  = tid & 31;
  const int wv    = tid >> 5;            // 0..31
  const int m     = lane & 15;           // A-row (batch) / B-row (col n)
  const int koff  = (lane >> 4) * 8;     // fragment K sub-offset
  const int half  = (lane >> 4) << 3;    // D rows: rv = v + half
  const int bbase = blockIdx.x * 16;
  const int col   = wv * 16 + m;         // hidden column owned by this lane

  const bf16_t* arow_x  = xstage + m * XST;
  const bf16_t* arow_h0 = hstate + m * HST;
  const bf16_t* arow_h1 = hstate + 16 * HST + m * HST;

  for (int i = tid; i < 2 * 16 * HST; i += 1024) hstate[i] = (bf16_t)0.0f;
  __syncthreads();

  v8f c_reg[2];
#pragma unroll
  for (int l = 0; l < 2; ++l)
#pragma unroll
    for (int v = 0; v < 8; ++v) c_reg[l][v] = 0.0f;

  for (int t = 0; t < Sk; ++t) {
    if (tid < 128) red[tid] = 0.0f;                       // both layers' LN accums
    for (int i = tid; i < 16 * DINk; i += 1024) {         // stage x_t as bf16
      int mm = i >> 8, cc = i & 255;
      xstage[mm * XST + cc] = (bf16_t)x[((size_t)(bbase + mm) * Sk + t) * DINk + cc];
    }
    __syncthreads();

#pragma unroll
    for (int l = 0; l < 2; ++l) {
      const int     K  = l ? 1024 : 768;
      const bf16_t* Wl = l ? Wcat1 : Wcat0;
      float*        rl = red + l * 64;

      v8f acc[4];
#pragma unroll
      for (int g2 = 0; g2 < 4; ++g2)
#pragma unroll
        for (int v = 0; v < 8; ++v) acc[g2][v] = 0.0f;

      if (l == 0) {
        gemm4_seg(arow_x,  koff, Wl, K, 0,    DINk >> 5, col, acc);  // x   : 8 chunks
        gemm4_seg(arow_h0, koff, Wl, K, DINk, Hk >> 5,   col, acc);  // h0  : 16 chunks
      } else {
        gemm4_seg(arow_h0, koff, Wl, K, 0,    Hk >> 5,   col, acc);  // h0  : 16 chunks
        gemm4_seg(arow_h1, koff, Wl, K, Hk,   Hk >> 5,   col, acc);  // h1  : 16 chunks
      }

      const float bi  = bcat[l * 2048 + 0 * Hk + col];
      const float bff = bcat[l * 2048 + 1 * Hk + col];
      const float bo  = bcat[l * 2048 + 2 * Hk + col];
      const float bg  = bcat[l * 2048 + 3 * Hk + col];

      float cnew[8], osav[8];
#pragma unroll
      for (int v = 0; v < 8; ++v) {
        float iv = sigmoidf_(acc[0][v] + bi);
        float fv = sigmoidf_(acc[1][v] + bff);
        float ov = sigmoidf_(acc[2][v] + bo);
        float gv = tanhf(acc[3][v] + bg);
        cnew[v]  = fv * c_reg[l][v] + iv * gv;
        osav[v]  = ov;
      }
      // per-batch-row partial sums for LN(c): reduce 16 cols across lanes
#pragma unroll
      for (int v = 0; v < 8; ++v) {
        float s = cnew[v], q = cnew[v] * cnew[v];
#pragma unroll
        for (int d = 1; d < 16; d <<= 1) { s += __shfl_xor(s, d, 32); q += __shfl_xor(q, d, 32); }
        if (m == 0) { atomicAdd(&rl[v + half], s); atomicAdd(&rl[16 + v + half], q); }
      }
      __syncthreads();

      const float gcv  = (l ? gc1 : gc0)[col];
      const float becv = (l ? bec1 : bec0)[col];
      float ph[8];
#pragma unroll
      for (int v = 0; v < 8; ++v) {
        int rv = v + half;
        float mean = rl[rv] * (1.0f / Hk);
        float var  = rl[16 + rv] * (1.0f / Hk) - mean * mean;
        float rstd = rsqrtf(var + 1e-5f);
        float cn   = (cnew[v] - mean) * rstd * gcv + becv;
        c_reg[l][v] = cn;
        ph[v] = osav[v] * tanhf(cn);
      }
#pragma unroll
      for (int v = 0; v < 8; ++v) {
        float s = ph[v], q = ph[v] * ph[v];
#pragma unroll
        for (int d = 1; d < 16; d <<= 1) { s += __shfl_xor(s, d, 32); q += __shfl_xor(q, d, 32); }
        if (m == 0) { atomicAdd(&rl[32 + v + half], s); atomicAdd(&rl[48 + v + half], q); }
      }
      __syncthreads();

      const float ghv  = (l ? gh1 : gh0)[col];
      const float behv = (l ? beh1 : beh0)[col];
#pragma unroll
      for (int v = 0; v < 8; ++v) {
        int rv = v + half;
        float mean = rl[32 + rv] * (1.0f / Hk);
        float var  = rl[48 + rv] * (1.0f / Hk) - mean * mean;
        float rstd = rsqrtf(var + 1e-5f);
        float hn   = (ph[v] - mean) * rstd * ghv + behv;
        hstate[l * 16 * HST + rv * HST + col] = (bf16_t)hn;
        if (l == 1)
          lstm_pad[((size_t)(bbase + rv) * SP + (t + 4)) * Hk + col] = (bf16_t)hn;
        if (t == Sk - 1) {
          hc_out[(size_t)l * Bk * Hk + (size_t)(bbase + rv) * Hk + col] = hn;
          hc_out[(size_t)2 * Bk * Hk + (size_t)l * Bk * Hk +
                 (size_t)(bbase + rv) * Hk + col] = c_reg[l][v];
        }
      }
      __syncthreads();
    }
  }
}

// ---------------------------------------------------------------------------
// Dilated conv as WMMA GEMM: out[b,t,o] = relu(b[o] + sum_k sum_c w[o,c,k]*in[b,t+(k-1)d,c])
// 8 waves/block, 1 16x16 tile per wave; padded time axis makes all taps in-bounds.
// ---------------------------------------------------------------------------
__global__ __launch_bounds__(256) void conv_kernel(
    const bf16_t* __restrict__ inp,  // lstm_pad [B][SP][H]
    const bf16_t* __restrict__ wr,   // [3][512][512]  (wr[k][o][c])
    const float*  __restrict__ bias,
    bf16_t* __restrict__ outp,       // [B][S][H] bf16 (relu applied)
    int dil)
{
  const int lane = threadIdx.x & 31;
  const int wv   = threadIdx.x >> 5;
  const int tile = blockIdx.x * 8 + wv;        // 32768 tiles exactly
  const int nt   = tile & 31, mt = tile >> 5;
  const int b    = mt >> 5;
  const int t0   = (mt & 31) << 4;
  const int o0   = nt << 4;
  const int m    = lane & 15;
  const int koff = (lane >> 4) * 8;
  const int half = (lane >> 4) << 3;

  v8f acc;
#pragma unroll
  for (int v = 0; v < 8; ++v) acc[v] = 0.0f;

  for (int k = 0; k < 3; ++k) {
    const bf16_t* arow = inp + ((size_t)b * SP + 4 + t0 + m + (k - 1) * dil) * Hk;
    const bf16_t* brow = wr + (size_t)k * Hk * Hk + (size_t)(o0 + m) * Hk;
#pragma unroll
    for (int kc = 0; kc < 16; ++kc) {
      int kb = (kc << 5) + koff;
      __builtin_prefetch(brow + kb + 32, 0, 0);
      v16bf a  = load_frag16(arow, kb);
      v16bf bf = load_frag16(brow, kb);
      acc = __builtin_amdgcn_wmma_f32_16x16x32_bf16(false, a, false, bf, (short)0,
                                                    acc, false, false);
    }
  }
  const float bo = bias[o0 + m];
#pragma unroll
  for (int v = 0; v < 8; ++v) {
    int rv = v + half;
    float val = acc[v] + bo;
    val = val > 0.0f ? val : 0.0f;
    outp[((size_t)b * Sk + t0 + rv) * Hk + o0 + m] = (bf16_t)val;
  }
}

// ---------------------------------------------------------------------------
// Projection: temporal = concat(lstm, conv0, conv1, conv2) @ proj_w.T + b
// K-loop runs over the 4 source segments directly (no concat materialized).
// ---------------------------------------------------------------------------
__global__ __launch_bounds__(256) void proj_kernel(
    const bf16_t* __restrict__ lstm_pad,
    const bf16_t* __restrict__ cv0,
    const bf16_t* __restrict__ cv1,
    const bf16_t* __restrict__ cv2,
    const bf16_t* __restrict__ projw,  // [512][2048] bf16, native [N][K]
    const float*  __restrict__ pb,
    float* __restrict__ outp)          // [B][S][H] fp32
{
  const int lane = threadIdx.x & 31;
  const int wv   = threadIdx.x >> 5;
  const int tile = blockIdx.x * 8 + wv;
  const int nt   = tile & 31, mt = tile >> 5;
  const int b    = mt >> 5;
  const int t0   = (mt & 31) << 4;
  const int n0   = nt << 4;
  const int m    = lane & 15;
  const int koff = (lane >> 4) * 8;
  const int half = (lane >> 4) << 3;

  v8f acc;
#pragma unroll
  for (int v = 0; v < 8; ++v) acc[v] = 0.0f;

  const bf16_t* arows[4];
  arows[0] = lstm_pad + ((size_t)b * SP + 4 + t0 + m) * Hk;
  arows[1] = cv0 + ((size_t)b * Sk + t0 + m) * Hk;
  arows[2] = cv1 + ((size_t)b * Sk + t0 + m) * Hk;
  arows[3] = cv2 + ((size_t)b * Sk + t0 + m) * Hk;

#pragma unroll
  for (int s = 0; s < 4; ++s) {
    const bf16_t* arow = arows[s];
    const bf16_t* brow = projw + (size_t)(n0 + m) * 2048 + s * Hk;
#pragma unroll
    for (int kc = 0; kc < 16; ++kc) {
      int kb = (kc << 5) + koff;
      __builtin_prefetch(brow + kb + 32, 0, 0);
      v16bf a  = load_frag16(arow, kb);
      v16bf bf = load_frag16(brow, kb);
      acc = __builtin_amdgcn_wmma_f32_16x16x32_bf16(false, a, false, bf, (short)0,
                                                    acc, false, false);
    }
  }
  const float bo = pb[n0 + m];
#pragma unroll
  for (int v = 0; v < 8; ++v) {
    int rv = v + half;
    outp[((size_t)b * Sk + t0 + rv) * Hk + n0 + m] = acc[v] + bo;
  }
}

// ---------------------------------------------------------------------------
extern "C" void kernel_launch(void* const* d_in, const int* in_sizes, int n_in,
                              void* d_out, int out_size, void* d_ws, size_t ws_size,
                              hipStream_t stream) {
  const float* x   = (const float*)d_in[0];
  const float* Wi0 = (const float*)d_in[1],  *bi0 = (const float*)d_in[2];
  const float* Wf0 = (const float*)d_in[3],  *bf0 = (const float*)d_in[4];
  const float* Wo0 = (const float*)d_in[5],  *bo0 = (const float*)d_in[6];
  const float* Wg0 = (const float*)d_in[7],  *bg0 = (const float*)d_in[8];
  const float* gc0 = (const float*)d_in[9],  *bec0 = (const float*)d_in[10];
  const float* gh0 = (const float*)d_in[11], *beh0 = (const float*)d_in[12];
  const float* Wi1 = (const float*)d_in[13], *bi1 = (const float*)d_in[14];
  const float* Wf1 = (const float*)d_in[15], *bf1 = (const float*)d_in[16];
  const float* Wo1 = (const float*)d_in[17], *bo1 = (const float*)d_in[18];
  const float* Wg1 = (const float*)d_in[19], *bg1 = (const float*)d_in[20];
  const float* gc1 = (const float*)d_in[21], *bec1 = (const float*)d_in[22];
  const float* gh1 = (const float*)d_in[23], *beh1 = (const float*)d_in[24];
  const float* cw0 = (const float*)d_in[25], *cb0 = (const float*)d_in[26];
  const float* cw1 = (const float*)d_in[27], *cb1 = (const float*)d_in[28];
  const float* cw2 = (const float*)d_in[29], *cb2 = (const float*)d_in[30];
  // d_in[31..34]: attention params -> dead code in reference, ignored.
  const float* pw  = (const float*)d_in[35], *pbias = (const float*)d_in[36];

  char* wsp = (char*)d_ws;
  auto take = [&](size_t bytes) -> char* {
    char* p = wsp; wsp += (bytes + 255) & ~(size_t)255; return p;
  };
  bf16_t* Wcat0 = (bf16_t*)take((size_t)2048 * 768 * 2);
  bf16_t* Wcat1 = (bf16_t*)take((size_t)2048 * 1024 * 2);
  float*  bcat  = (float*) take((size_t)2 * 2048 * 4);
  bf16_t* cwr0  = (bf16_t*)take((size_t)3 * 512 * 512 * 2);
  bf16_t* cwr1  = (bf16_t*)take((size_t)3 * 512 * 512 * 2);
  bf16_t* cwr2  = (bf16_t*)take((size_t)3 * 512 * 512 * 2);
  bf16_t* pwb   = (bf16_t*)take((size_t)512 * 2048 * 2);
  bf16_t* lpad  = (bf16_t*)take((size_t)Bk * SP * Hk * 2);
  bf16_t* cvo0  = (bf16_t*)take((size_t)Bk * Sk * Hk * 2);
  bf16_t* cvo1  = (bf16_t*)take((size_t)Bk * Sk * Hk * 2);
  bf16_t* cvo2  = (bf16_t*)take((size_t)Bk * Sk * Hk * 2);

  float* outT  = (float*)d_out;
  float* outHC = outT + (size_t)Bk * Sk * Hk;

  cvt_lstm_w<<<512, 256, 0, stream>>>(Wi0, Wf0, Wo0, Wg0, bi0, bf0, bo0, bg0,
                                      Wcat0, bcat, 768);
  cvt_lstm_w<<<512, 256, 0, stream>>>(Wi1, Wf1, Wo1, Wg1, bi1, bf1, bo1, bg1,
                                      Wcat1, bcat + 2048, 1024);
  cvt_conv_w<<<512, 256, 0, stream>>>(cw0, cwr0);
  cvt_conv_w<<<512, 256, 0, stream>>>(cw1, cwr1);
  cvt_conv_w<<<512, 256, 0, stream>>>(cw2, cwr2);
  cvt_f2bf  <<<512, 256, 0, stream>>>(pw, pwb, (size_t)512 * 2048);
  zero_bf16 <<<512, 256, 0, stream>>>(lpad, (size_t)Bk * SP * Hk);

  size_t smem = (size_t)(2 * 16 * HST + 16 * XST) * sizeof(bf16_t) + 128 * sizeof(float);
  lstm_kernel<<<2, 1024, smem, stream>>>(x, Wcat0, Wcat1, bcat,
                                         gc0, bec0, gh0, beh0,
                                         gc1, bec1, gh1, beh1,
                                         lpad, outHC);

  conv_kernel<<<4096, 256, 0, stream>>>(lpad, cwr0, cb0, cvo0, 1);
  conv_kernel<<<4096, 256, 0, stream>>>(lpad, cwr1, cb1, cvo1, 2);
  conv_kernel<<<4096, 256, 0, stream>>>(lpad, cwr2, cb2, cvo2, 4);

  proj_kernel<<<4096, 256, 0, stream>>>(lpad, cvo0, cvo1, cvo2, pwb, pbias, outT);
}
